// ParticleNetWrapper_89524298318158
// MI455X (gfx1250) — matrix operand order repaired and checked by reading, not compiled, not run on hardware
//
#include <hip/hip_runtime.h>

// ---------------------------------------------------------------------------
// Types / helpers for bf16 WMMA (CDNA5 / gfx1250, wave32)
// ---------------------------------------------------------------------------
typedef __attribute__((ext_vector_type(16))) __bf16 v16bf;
typedef __attribute__((ext_vector_type(8)))  float  v8f;

union BFrag { unsigned int u[8]; v16bf v; };

__device__ __forceinline__ unsigned short f2bf(float x) {
    unsigned int u = __float_as_uint(x);
    unsigned int r = u + 0x7FFFu + ((u >> 16) & 1u);   // round-to-nearest-even
    return (unsigned short)(r >> 16);
}

// A-fragment (16x32 bf16), row-major source with element stride `stride`.
// ISA 16-bit A layout: lane m=l%16, half=l/16; VGPR v holds K pair
// kb = (v&3)*2 + half*8 + (v>=4)*16 relative to k0.
__device__ __forceinline__ void load_a16(const unsigned short* base, int row0,
                                         int stride, int k0, BFrag& f) {
    const int l = threadIdx.x & 31;
    const int m = l & 15, half = l >> 4;
    const unsigned short* rp = base + (row0 + m) * stride + k0 + half * 8;
#pragma unroll
    for (int v = 0; v < 8; ++v) {
        int kb = ((v & 3) << 1) + ((v >> 2) << 4);
        f.u[v] = *(const unsigned int*)(rp + kb);
    }
}

// B-fragment from pre-packed weights: [(kt*ntiles + nt)*32 + lane]*8 dwords.
__device__ __forceinline__ void load_b16(const unsigned int* __restrict__ pack,
                                         int kt, int nt, int ntiles, BFrag& f) {
    const int l = threadIdx.x & 31;
    const unsigned int* p = pack + ((((kt * ntiles) + nt) * 32) + l) * 8;
#pragma unroll
    for (int v = 0; v < 8; ++v) f.u[v] = p[v];
}

// ---------------------------------------------------------------------------
// Workspace layout (bytes)
// ---------------------------------------------------------------------------
#define WS_COORDS0 0u            // 64*128*2 f32           = 65536
#define WS_XA      65536u        // 64*128*256 f32 max     = 8388608
#define WS_XB      8454144u      // 8388608
#define WS_IDX     16842752u     // 64*128*16 int          = 524288
#define WS_FUSED   17367040u     // 64*128*448 bf16        = 7340032
#define WS_POOLED  24707072u     // 64*384 f32             = 98304
#define WS_COUNTS  24805376u     // 64 f32                 = 256
#define WS_WPACK   24805632u     // 214016 u32             = 856064

// ---------------------------------------------------------------------------
// Small kernels
// ---------------------------------------------------------------------------
__global__ void pn_zero_kernel(float* p, int n) {
    int i = blockIdx.x * blockDim.x + threadIdx.x;
    if (i < n) p[i] = 0.f;
}

__global__ void pn_preprocess_kernel(const float* __restrict__ points,
                                     const float* __restrict__ features,
                                     const int* __restrict__ mask,
                                     const float* __restrict__ bns,
                                     const float* __restrict__ bnb,
                                     float* __restrict__ coords0,
                                     float* __restrict__ x0,
                                     float* __restrict__ counts) {
    int i = blockIdx.x * blockDim.x + threadIdx.x;
    if (i >= 64 * 128) return;
    int e = i >> 7, p = i & 127;
    int mv = mask[e * 128 + p];
    float m  = mv ? 1.f : 0.f;
    float sh = mv ? 0.f : 1.0e9f;
#pragma unroll
    for (int d = 0; d < 2; ++d)
        coords0[i * 2 + d] = points[(e * 2 + d) * 128 + p] * m + sh;
#pragma unroll
    for (int c = 0; c < 4; ++c) {
        float f = features[(e * 4 + c) * 128 + p];
        x0[i * 4 + c] = mv ? (f * bns[c] + bnb[c]) : 0.f;
    }
    if (mv) atomicAdd(&counts[e], 1.f);
}

// Pack f32 weight [K][O] (K zero-padded to Kpad) into WMMA B-fragment layout.
__global__ void pn_pack_w_kernel(const float* __restrict__ W, int K, int Kpad,
                                 int O, unsigned int* __restrict__ out) {
    int i = blockIdx.x * blockDim.x + threadIdx.x;
    int total = (Kpad * O) >> 1;
    if (i >= total) return;
    int v    = i & 7;
    int l    = (i >> 3) & 31;
    int rest = i >> 8;
    int ntiles = O >> 4;
    int nt = rest % ntiles;
    int kt = rest / ntiles;
    int half = l >> 4;
    int n  = nt * 16 + (l & 15);
    int k0 = kt * 32 + (v & 3) * 2 + half * 8 + (v >> 2) * 16;
    float f0 = (k0     < K) ? W[k0 * O + n]       : 0.f;
    float f1 = (k0 + 1 < K) ? W[(k0 + 1) * O + n] : 0.f;
    out[i] = (unsigned int)f2bf(f0) | ((unsigned int)f2bf(f1) << 16);
}

// kNN: one workgroup per event, one thread per point. Top-17 by
// (neg_dist desc, index asc), drop first — matches lax.top_k(...)[:,:,1:].
__global__ void pn_knn_kernel(const float* __restrict__ coords, int C,
                              const int* __restrict__ mask, int addShift,
                              int* __restrict__ idx) {
    extern __shared__ float sm_k[];
    int e = blockIdx.x;
    int p = threadIdx.x;          // 128 threads
    float sh = 0.f;
    if (addShift) sh = mask[e * 128 + p] ? 0.f : 1.0e9f;
    for (int c = 0; c < C; ++c)
        sm_k[p * C + c] = coords[(e * 128 + p) * C + c] + sh;
    __syncthreads();
    float vals[17]; int ids[17];
#pragma unroll
    for (int j = 0; j < 17; ++j) { vals[j] = -3.0e38f; ids[j] = 0; }
    for (int q = 0; q < 128; ++q) {
        float d = 0.f;
        for (int c = 0; c < C; ++c) {
            float t = sm_k[p * C + c] - sm_k[q * C + c];
            d += t * t;
        }
        float v = -d;
        if (v > vals[16]) {                 // strict > keeps ties index-ascending
            int j = 16;
            while (j > 0 && v > vals[j - 1]) {
                vals[j] = vals[j - 1]; ids[j] = ids[j - 1]; --j;
            }
            vals[j] = v; ids[j] = q;
        }
    }
#pragma unroll
    for (int k = 0; k < 16; ++k)
        idx[(e * 128 + p) * 16 + k] = ids[k + 1];
}

// ---------------------------------------------------------------------------
// GEMM layer: [128 x K] @ [K x COUT] -> relu(*s+b) -> bf16 dst in LDS.
// Wave owns row-tile rt == wave. Column tiles processed in chunks of <=8 so
// live accumulators stay <= 64 VGPRs (no spills); A loaded once per kt per
// chunk from LDS; B streams from L2-resident packed weights.
// ---------------------------------------------------------------------------
template <int COUT>
__device__ __forceinline__ void pn_gemm_layer(const unsigned short* src, int srcStride,
                                              int KT, const unsigned int* __restrict__ w,
                                              const float* __restrict__ s,
                                              const float* __restrict__ b,
                                              unsigned short* dst) {
    const int tid  = threadIdx.x;
    const int rt   = tid >> 5;                 // wave id == row tile
    const int lane = tid & 31;
    const int nIdx = lane & 15, half = lane >> 4;
    constexpr int NT = COUT / 16;
    constexpr int CH = (NT < 8) ? NT : 8;
    for (int c0 = 0; c0 < NT; c0 += CH) {
        v8f acc[CH];
#pragma unroll
        for (int j = 0; j < CH; ++j)
            acc[j] = (v8f){0.f, 0.f, 0.f, 0.f, 0.f, 0.f, 0.f, 0.f};
        for (int kt = 0; kt < KT; ++kt) {
            BFrag a;
            load_a16(src, rt * 16, srcStride, kt * 32, a);
#pragma unroll
            for (int j = 0; j < CH; ++j) {
                BFrag bb;
                load_b16(w, kt, c0 + j, NT, bb);
                acc[j] = __builtin_amdgcn_wmma_f32_16x16x32_bf16(
                    false, a.v, false, bb.v, (short)0, acc[j], false, false);
            }
        }
#pragma unroll
        for (int j = 0; j < CH; ++j) {
            int ct = c0 + j;
            float sv = s[ct * 16 + nIdx], bv = b[ct * 16 + nIdx];
#pragma unroll
            for (int r = 0; r < 8; ++r) {
                float vv = fmaxf(acc[j][r] * sv + bv, 0.f);
                dst[(rt * 16 + r + half * 8) * COUT + ct * 16 + nIdx] = f2bf(vv);
            }
        }
    }
}

// ---------------------------------------------------------------------------
// Fused EdgeConv block: gather edges -> conv1 -> conv2 -> conv3 -> mean(k)
// -> shortcut -> relu -> mask.  One WG per (event, 8-point group).
// ---------------------------------------------------------------------------
template <int CIN, int COUT, int K1PAD>
__global__ void __launch_bounds__(256, 1)
pn_edgeconv_kernel(const float* __restrict__ xin,
                   float* __restrict__ xout,
                   unsigned short* __restrict__ fused, int fusedOff,
                   const int* __restrict__ idx,
                   const int* __restrict__ mask,
                   const float* __restrict__ scW,
                   const float* __restrict__ scs,
                   const float* __restrict__ scb,
                   const float* __restrict__ s1, const float* __restrict__ b1,
                   const float* __restrict__ s2, const float* __restrict__ b2,
                   const float* __restrict__ s3, const float* __restrict__ b3,
                   const unsigned int* __restrict__ w1,
                   const unsigned int* __restrict__ w2,
                   const unsigned int* __restrict__ w3) {
    extern __shared__ unsigned short sm_ec[];
    unsigned short* E  = sm_ec;                 // 128 * K1PAD  (edge features)
    unsigned short* X1 = E  + 128 * K1PAD;      // 128 * COUT
    unsigned short* X2 = X1 + 128 * COUT;       // 128 * COUT
    const int e     = blockIdx.x >> 4;
    const int pBase = (blockIdx.x & 15) * 8;
    const int tid   = threadIdx.x;
    const int CIN2  = 2 * CIN;

    // Stage edge features [ctr | nbr-ctr] (zero-padded to K1PAD) into LDS.
    for (int i = tid; i < 128 * K1PAD; i += 256) {
        int row = i / K1PAD, c = i % K1PAD;
        int pt = row >> 4, kk = row & 15;
        int p = pBase + pt;
        float v = 0.f;
        if (c < CIN) {
            v = xin[(e * 128 + p) * CIN + c];
        } else if (c < CIN2) {
            int cc = c - CIN;
            int nb = idx[(e * 128 + p) * 16 + kk];
            v = xin[(e * 128 + nb) * CIN + cc] - xin[(e * 128 + p) * CIN + cc];
        }
        E[i] = f2bf(v);
    }
    __syncthreads();

    pn_gemm_layer<COUT>(E, K1PAD, K1PAD / 32, w1, s1, b1, X1);
    __syncthreads();
    pn_gemm_layer<COUT>(X1, COUT, COUT / 32, w2, s2, b2, X2);
    __syncthreads();

    // conv3 + mean over k + shortcut + relu + mask (row-tile == one point's k edges)
    const int rt   = tid >> 5;                 // wave id == row tile == local point
    const int lane = tid & 31;
    const int nIdx = lane & 15, half = lane >> 4;
    constexpr int NT = COUT / 16;
    constexpr int CH = (NT < 8) ? NT : 8;
    for (int c0 = 0; c0 < NT; c0 += CH) {
        v8f acc[CH];
#pragma unroll
        for (int j = 0; j < CH; ++j)
            acc[j] = (v8f){0.f, 0.f, 0.f, 0.f, 0.f, 0.f, 0.f, 0.f};
        for (int kt = 0; kt < COUT / 32; ++kt) {
            BFrag a;
            load_a16(X2, rt * 16, COUT, kt * 32, a);
#pragma unroll
            for (int j = 0; j < CH; ++j) {
                BFrag bb;
                load_b16(w3, kt, c0 + j, NT, bb);
                acc[j] = __builtin_amdgcn_wmma_f32_16x16x32_bf16(
                    false, a.v, false, bb.v, (short)0, acc[j], false, false);
            }
        }
#pragma unroll
        for (int j = 0; j < CH; ++j) {
            int ct = c0 + j;
            float sv = s3[ct * 16 + nIdx], bv = b3[ct * 16 + nIdx];
            float sum = 0.f;
#pragma unroll
            for (int r = 0; r < 8; ++r) sum += fmaxf(acc[j][r] * sv + bv, 0.f);
            sum += __shfl_xor(sum, 16, 32);    // lanes l, l^16 share a column
            if (half == 0) {
                int p = pBase + rt;
                int o = ct * 16 + nIdx;
                float mv = mask[e * 128 + p] ? 1.f : 0.f;
                float sc = 0.f;
                for (int c = 0; c < CIN; ++c)
                    sc += xin[(e * 128 + p) * CIN + c] * scW[c * COUT + o];
                sc = sc * scs[o] + scb[o];
                float ov = fmaxf(sc + sum * 0.0625f, 0.f) * mv;
                xout[(e * 128 + p) * COUT + o] = ov;
                fused[(e * 128 + p) * 448 + fusedOff + o] = f2bf(ov);
            }
        }
    }
}

// ---------------------------------------------------------------------------
// Fusion GEMM (448 -> 384) + relu + mask + masked sum over points (atomic)
// ---------------------------------------------------------------------------
__global__ void __launch_bounds__(256, 1)
pn_fuse_kernel(const unsigned short* __restrict__ fused,
               const int* __restrict__ mask,
               const float* __restrict__ fuss,
               const float* __restrict__ fusb,
               const unsigned int* __restrict__ wf,
               float* __restrict__ pooled) {
    const int e    = blockIdx.x;
    const int tid  = threadIdx.x;
    const int rt   = tid >> 5;                 // wave id == row tile
    const int lane = tid & 31;
    const int nIdx = lane & 15, half = lane >> 4;
    const unsigned short* A = fused + (size_t)e * 128 * 448;
    constexpr int NT = 24;                     // 384 / 16
    constexpr int CH = 8;
    for (int c0 = 0; c0 < NT; c0 += CH) {
        v8f acc[CH];
#pragma unroll
        for (int j = 0; j < CH; ++j)
            acc[j] = (v8f){0.f, 0.f, 0.f, 0.f, 0.f, 0.f, 0.f, 0.f};
        for (int kt = 0; kt < 448 / 32; ++kt) {
            BFrag a;
            load_a16(A, rt * 16, 448, kt * 32, a);
#pragma unroll
            for (int j = 0; j < CH; ++j) {
                BFrag bb;
                load_b16(wf, kt, c0 + j, NT, bb);
                acc[j] = __builtin_amdgcn_wmma_f32_16x16x32_bf16(
                    false, a.v, false, bb.v, (short)0, acc[j], false, false);
            }
        }
#pragma unroll
        for (int j = 0; j < CH; ++j) {
            int ct = c0 + j;
            float sv = fuss[ct * 16 + nIdx], bv = fusb[ct * 16 + nIdx];
            float sum = 0.f;
#pragma unroll
            for (int r = 0; r < 8; ++r) {
                int row = rt * 16 + r + half * 8;
                float mv = mask[e * 128 + row] ? 1.f : 0.f;
                sum += fmaxf(acc[j][r] * sv + bv, 0.f) * mv;
            }
            sum += __shfl_xor(sum, 16, 32);
            if (half == 0)
                atomicAdd(&pooled[e * 384 + ct * 16 + nIdx], sum);
        }
    }
}

// ---------------------------------------------------------------------------
// Head MLP (tiny, f32 scalar, single workgroup, LDS-staged)
// ---------------------------------------------------------------------------
__global__ void pn_head_kernel(const float* __restrict__ pooled,
                               const float* __restrict__ counts,
                               const float* __restrict__ fcW, const float* __restrict__ fcb,
                               const float* __restrict__ h1W, const float* __restrict__ h1b,
                               const float* __restrict__ h2W, const float* __restrict__ h2b,
                               const float* __restrict__ h3W, const float* __restrict__ h3b,
                               float* __restrict__ out) {
    extern __shared__ float sm_h[];
    float* P = sm_h;             // 64*384
    float* A = P + 64 * 384;     // 64*256
    float* B = A + 64 * 256;     // 64*128
    float* C = B + 64 * 128;     // 64*64
    const int tid = threadIdx.x;
    for (int i = tid; i < 64 * 384; i += 256) {
        int r = i / 384;
        P[i] = pooled[i] / fmaxf(counts[r], 1.f);
    }
    __syncthreads();
    for (int i = tid; i < 64 * 256; i += 256) {
        int r = i / 256, o = i % 256;
        float acc = fcb[o];
        for (int c = 0; c < 384; ++c) acc += P[r * 384 + c] * fcW[c * 256 + o];
        A[i] = fmaxf(acc, 0.f);
    }
    __syncthreads();
    for (int i = tid; i < 64 * 128; i += 256) {
        int r = i / 128, o = i % 128;
        float acc = h1b[o];
        for (int c = 0; c < 256; ++c) acc += A[r * 256 + c] * h1W[c * 128 + o];
        B[i] = acc > 0.f ? acc : 0.01f * acc;
    }
    __syncthreads();
    for (int i = tid; i < 64 * 64; i += 256) {
        int r = i / 64, o = i % 64;
        float acc = h2b[o];
        for (int c = 0; c < 128; ++c) acc += B[r * 128 + c] * h2W[c * 64 + o];
        C[i] = acc > 0.f ? acc : 0.01f * acc;
    }
    __syncthreads();
    for (int i = tid; i < 128; i += 256) {
        int r = i >> 1, o = i & 1;
        float acc = h3b[o];
        for (int c = 0; c < 64; ++c) acc += C[r * 64 + c] * h3W[c * 2 + o];
        out[i] = acc;     // flat (8,8,2) == flat (8,2,8) view
    }
}

// ---------------------------------------------------------------------------
// Launch
// ---------------------------------------------------------------------------
extern "C" void kernel_launch(void* const* d_in, const int* in_sizes, int n_in,
                              void* d_out, int out_size, void* d_ws, size_t ws_size,
                              hipStream_t stream) {
    (void)in_sizes; (void)n_in; (void)out_size; (void)ws_size;
    const float* points   = (const float*)d_in[0];
    const float* features = (const float*)d_in[1];
    const int*   mask     = (const int*)d_in[2];
    const float* bns      = (const float*)d_in[3];
    const float* bnb      = (const float*)d_in[4];

    char* ws = (char*)d_ws;
    float*          coords0 = (float*)(ws + WS_COORDS0);
    float*          xA      = (float*)(ws + WS_XA);
    float*          xB      = (float*)(ws + WS_XB);
    int*            idx     = (int*)(ws + WS_IDX);
    unsigned short* fused   = (unsigned short*)(ws + WS_FUSED);
    float*          pooled  = (float*)(ws + WS_POOLED);
    float*          counts  = (float*)(ws + WS_COUNTS);
    unsigned int*   wpack   = (unsigned int*)(ws + WS_WPACK);

    // zero pooled (24576) + counts (64), contiguous
    pn_zero_kernel<<<(24640 + 255) / 256, 256, 0, stream>>>(pooled, 24640);
    pn_preprocess_kernel<<<32, 256, 0, stream>>>(points, features, mask, bns, bnb,
                                                 coords0, xA, counts);
    // pack all GEMM weights into B-fragment layout (bf16)
    struct PW { int in; int K; int Kp; int O; int off; };
    const PW pws[10] = {
        {5,   8,  32,  64,      0}, {8,  64,  64,  64,   1024}, {11, 64,  64,  64,   3072},
        {17, 128, 128, 128,  5120}, {20, 128, 128, 128, 13312}, {23, 128, 128, 128, 21504},
        {29, 256, 256, 256, 29696}, {32, 256, 256, 256, 62464}, {35, 256, 256, 256, 95232},
        {41, 448, 448, 384, 128000}};
    for (int i = 0; i < 10; ++i) {
        int total = (pws[i].Kp * pws[i].O) >> 1;
        pn_pack_w_kernel<<<(total + 255) / 256, 256, 0, stream>>>(
            (const float*)d_in[pws[i].in], pws[i].K, pws[i].Kp, pws[i].O,
            wpack + pws[i].off);
    }

    // block 0: coords dim 2 (pre-shifted), 4 -> 64
    pn_knn_kernel<<<64, 128, 128 * 2 * 4, stream>>>(coords0, 2, mask, 0, idx);
    pn_edgeconv_kernel<4, 64, 32><<<1024, 256, 40960, stream>>>(
        xA, xB, fused, 0, idx, mask,
        (const float*)d_in[14], (const float*)d_in[15], (const float*)d_in[16],
        (const float*)d_in[6],  (const float*)d_in[7],
        (const float*)d_in[9],  (const float*)d_in[10],
        (const float*)d_in[12], (const float*)d_in[13],
        wpack + 0, wpack + 1024, wpack + 3072);

    // block 1: coords = x (64 ch), 64 -> 128
    pn_knn_kernel<<<64, 128, 128 * 64 * 4, stream>>>(xB, 64, mask, 1, idx);
    pn_edgeconv_kernel<64, 128, 128><<<1024, 256, 98304, stream>>>(
        xB, xA, fused, 64, idx, mask,
        (const float*)d_in[26], (const float*)d_in[27], (const float*)d_in[28],
        (const float*)d_in[18], (const float*)d_in[19],
        (const float*)d_in[21], (const float*)d_in[22],
        (const float*)d_in[24], (const float*)d_in[25],
        wpack + 5120, wpack + 13312, wpack + 21504);

    // block 2: coords = x (128 ch), 128 -> 256
    pn_knn_kernel<<<64, 128, 128 * 128 * 4, stream>>>(xA, 128, mask, 1, idx);
    pn_edgeconv_kernel<128, 256, 256><<<1024, 256, 196608, stream>>>(
        xA, xB, fused, 192, idx, mask,
        (const float*)d_in[38], (const float*)d_in[39], (const float*)d_in[40],
        (const float*)d_in[30], (const float*)d_in[31],
        (const float*)d_in[33], (const float*)d_in[34],
        (const float*)d_in[36], (const float*)d_in[37],
        wpack + 29696, wpack + 62464, wpack + 95232);

    // fusion + masked pooling
    pn_fuse_kernel<<<64, 256, 0, stream>>>(fused, mask,
        (const float*)d_in[42], (const float*)d_in[43], wpack + 128000, pooled);

    // head MLP
    pn_head_kernel<<<1, 256, 212992, stream>>>(pooled, counts,
        (const float*)d_in[44], (const float*)d_in[45],
        (const float*)d_in[46], (const float*)d_in[47],
        (const float*)d_in[48], (const float*)d_in[49],
        (const float*)d_in[50], (const float*)d_in[51],
        (float*)d_out);
}
